// RingMultiheadDilatedAttentionHybrid_23029614641319
// MI455X (gfx1250) — matrix operand order repaired and checked by reading, not compiled, not run on hardware
//
#include <hip/hip_runtime.h>
#include <hip/hip_bf16.h>
#include <stdint.h>

// ---------------------------------------------------------------------------
// Problem constants (match reference)
// ---------------------------------------------------------------------------
#define B_  2
#define N_  4096
#define E_  1024
#define H_  16
#define D_  64
#define M_  (B_ * N_)          // 8192 flattened tokens

typedef __bf16 bf16;
typedef __attribute__((ext_vector_type(16))) __bf16 v16bf;
typedef __attribute__((ext_vector_type(8)))  float  v8f;

// ---------------------------------------------------------------------------
// WMMA helper: D = A(16x32 bf16) * B(32x16 bf16) + C(16x16 f32)
// ---------------------------------------------------------------------------
__device__ __forceinline__ v8f wmma_bf16(v16bf a, v16bf b, v8f c) {
  return __builtin_amdgcn_wmma_f32_16x16x32_bf16(
      /*neg_a=*/false, a, /*neg_b=*/false, b,
      /*c_mod=*/(short)0, c, /*reuse_a=*/false, /*reuse_b=*/false);
}

// A-matrix fragment (16x32, MxK). Contiguous k order (same as load_b_frag):
//   lane m = lane&15;  lanes 0-15: k = 0..15; lanes 16-31: k = 16..31.
__device__ __forceinline__ v16bf load_a_frag(const bf16* base, int stride) {
  const int lane = threadIdx.x & 31;
  const int m    = lane & 15;
  const int hi   = (lane >> 4) << 4;                 // 0 or 16
  const bf16* row = base + (size_t)m * stride;
  v16bf a;
#pragma unroll
  for (int j = 0; j < 8; ++j) {
    a[j]     = row[hi + j];
    a[8 + j] = row[8 + hi + j];
  }
  return a;
}

// B-matrix fragment (32x16, KxN), stored n-major (storage[n][k], k contiguous).
//   lane n = lane&15; lanes 0-15 hold k=0..15, lanes 16-31 hold k=16..31.
__device__ __forceinline__ v16bf load_b_frag(const bf16* base, int stride) {
  const int lane = threadIdx.x & 31;
  const int n    = lane & 15;
  const int kk   = (lane >> 4) << 4;                 // 0 or 16
  const bf16* p = base + (size_t)n * stride + kk;
  v16bf b;
#pragma unroll
  for (int j = 0; j < 16; ++j) b[j] = p[j];
  return b;
}

// ---------------------------------------------------------------------------
// Async global->LDS copy of 32 bytes per lane (pure bf16 tile copy; no VGPR
// round trip, tracked on ASYNCcnt instead of LOADcnt+DScnt). The ISA adds
// INST_OFFSET to BOTH the global and the LDS address, so one base covers the
// second 16-byte half via offset:16. Low 32 bits of a generic LDS pointer are
// the LDS byte offset (flat aperture places the base in addr[63:32]).
// ---------------------------------------------------------------------------
__device__ __forceinline__ void async_copy32(void* lds_dst, const void* gsrc) {
  unsigned           ldsoff = (unsigned)(uintptr_t)lds_dst;
  unsigned long long gaddr  = (unsigned long long)(uintptr_t)gsrc;
  asm volatile("global_load_async_to_lds_b128 %0, %1, off"
               :: "v"(ldsoff), "v"(gaddr) : "memory");
  asm volatile("global_load_async_to_lds_b128 %0, %1, off offset:16"
               :: "v"(ldsoff), "v"(gaddr) : "memory");
}
__device__ __forceinline__ void async_drain() {
  asm volatile("s_wait_asynccnt 0" ::: "memory");
}

// ---------------------------------------------------------------------------
// Weight convert: W[k][n] f32 (KxN row-major) -> WT[n][k] bf16 (NxK row-major)
// ---------------------------------------------------------------------------
__global__ void wconvert(const float* __restrict__ W, bf16* __restrict__ WT,
                         int K, int Nc) {
  int i = blockIdx.x * blockDim.x + threadIdx.x;
  if (i >= K * Nc) return;
  int n = i % Nc;
  int k = i / Nc;                                    // coalesced read of W
  WT[(size_t)n * K + k] = (bf16)W[i];
}

__global__ void zerof(float* __restrict__ p, int n) {
  int i = blockIdx.x * blockDim.x + threadIdx.x;
  if (i < n) p[i] = 0.0f;
}

// ---------------------------------------------------------------------------
// Tiled WMMA GEMM:  Y = f(X @ W + bias)
//   mode 0: Ybf  = bf16(acc + bias)            (q/k/v projections)
//   mode 1: Yf   = sigmoid(acc + bias)         (gate)
//   mode 2: X' = X*Gx; Yf = acc + bias  (f32)  (final output projection)
// Block tile 128x128, K-step 32, 256 threads = 8 waves, wave tile 32x64.
// B tile is a pure bf16 copy -> async global->LDS; A tile needs f32->bf16
// conversion (and gate fusion) so it goes through VGPRs.
// ---------------------------------------------------------------------------
#define BM 128
#define BN 128
#define BK 32
#define GPAD 8

__global__ __launch_bounds__(256) void gemm_bf16(
    const float* __restrict__ X, const float* __restrict__ Gx,
    const bf16* __restrict__ WT, const float* __restrict__ bias,
    bf16* __restrict__ Ybf, float* __restrict__ Yf,
    int M, int K, int Nc, int mode) {
  __shared__ bf16 sA[BM][BK + GPAD];                 // row pitch 80 B (16B mult)
  __shared__ bf16 sB[BN][BK + GPAD];

  const int tid  = threadIdx.x;
  const int lane = tid & 31;
  const int wave = tid >> 5;                         // 0..7
  const int wm   = wave >> 1;                        // 0..3 -> 32-row tile
  const int wn   = wave & 1;                         // 0..1 -> 64-col tile
  const int bm   = blockIdx.x * BM;
  const int bn   = blockIdx.y * BN;

  v8f acc[2][4];
#pragma unroll
  for (int mi = 0; mi < 2; ++mi)
#pragma unroll
    for (int nj = 0; nj < 4; ++nj)
#pragma unroll
      for (int j = 0; j < 8; ++j) acc[mi][nj][j] = 0.0f;

  // cooperative loader mapping: 256 threads cover 128 rows x 32 cols, 16/thread
  const int lr = tid >> 1;                           // 0..127 row
  const int lc = (tid & 1) * 16;                     // col base 0/16

  for (int k0 = 0; k0 < K; k0 += BK) {
    // --- B tile: async pure-copy of pre-transposed bf16 weights ---
    async_copy32(&sB[lr][lc], WT + (size_t)(bn + lr) * K + k0 + lc);
    // --- A tile: f32 -> bf16 (with optional gate fusion) ---
    {
      const float* xp = X + (size_t)(bm + lr) * K + k0 + lc;
      float xv[16];
#pragma unroll
      for (int e = 0; e < 16; ++e) xv[e] = xp[e];
      if (mode == 2) {
        const float* gp = Gx + (size_t)(bm + lr) * K + k0 + lc;
#pragma unroll
        for (int e = 0; e < 16; ++e) xv[e] *= gp[e];
      }
#pragma unroll
      for (int e = 0; e < 16; ++e) sA[lr][lc + e] = (bf16)xv[e];
    }
    async_drain();                                   // our B-copy done in LDS
    __syncthreads();

    v16bf af0 = load_a_frag(&sA[wm * 32][0], BK + GPAD);
    v16bf af1 = load_a_frag(&sA[wm * 32 + 16][0], BK + GPAD);
#pragma unroll
    for (int nj = 0; nj < 4; ++nj) {
      v16bf bf = load_b_frag(&sB[wn * 64 + nj * 16][0], BK + GPAD);
      acc[0][nj] = wmma_bf16(af0, bf, acc[0][nj]);
      acc[1][nj] = wmma_bf16(af1, bf, acc[1][nj]);
    }
    __syncthreads();
  }

  // epilogue: C layout m = j + 8*(lane>=16), n = lane&15
  const int half = lane >> 4;
  const int cn   = lane & 15;
#pragma unroll
  for (int mi = 0; mi < 2; ++mi)
#pragma unroll
    for (int nj = 0; nj < 4; ++nj)
#pragma unroll
      for (int j = 0; j < 8; ++j) {
        int row = bm + wm * 32 + mi * 16 + j + half * 8;
        int col = bn + wn * 64 + nj * 16 + cn;
        float v = acc[mi][nj][j] + bias[col];
        size_t idx = (size_t)row * Nc + col;
        if (mode == 0)      Ybf[idx] = (bf16)v;
        else if (mode == 1) Yf[idx]  = 1.0f / (1.0f + __expf(-v));
        else                Yf[idx]  = v;
      }
}

// ---------------------------------------------------------------------------
// Dilated flash attention. One block = 128 threads = 4 waves; each wave owns a
// 16-query x 64-d output tile of one (batch, segment, head). All waves share
// the K/V LDS tiles. s' = seg_len/rate = 1024 for both groups.
// K tile is a pure bf16 copy -> async global->LDS; V needs a transpose so it
// goes through VGPRs.
// ---------------------------------------------------------------------------
#define APAD 8

__global__ __launch_bounds__(128) void attn_kernel(
    const bf16* __restrict__ qb, const bf16* __restrict__ kb,
    const bf16* __restrict__ vb, float* __restrict__ attn,
    int seg_len, int rate, int off, int head0, int nseg) {
  __shared__ bf16 sK[32][64 + APAD];                 // keys x d, pitch 144 B
  __shared__ bf16 sVT[64][32 + APAD];                // d x keys  (B-frag ready)
  __shared__ bf16 sP[4][16][32 + APAD];              // per-wave P stash

  const int tid  = threadIdx.x;
  const int lane = tid & 31;
  const int wave = tid >> 5;

  int bid = blockIdx.x;
  const int qt  = bid & 15; bid >>= 4;               // 16 qtiles of 64 (s'=1024)
  const int hh  = bid & 7;  bid >>= 3;               // head within group (g=8)
  const int seg = bid % nseg;
  const int b   = bid / nseg;
  const int h   = head0 + hh;

  const int sp    = 1024;                            // s' (both groups)
  const int qbase = qt * 64 + wave * 16;

  // Preload Q fragments (held for the whole kernel). Dilated queries are a
  // uniform stride of rate*E elements, so load_a_frag applies directly.
  const size_t qtok0 = (size_t)b * N_ + (size_t)seg * seg_len + off +
                       (size_t)qbase * rate;
  const bf16* qptr = qb + qtok0 * E_ + h * D_;
  v16bf qf0 = load_a_frag(qptr, rate * E_);
  v16bf qf1 = load_a_frag(qptr + 32, rate * E_);

  v8f o[4];
  float mrun[8], lrun[8];
#pragma unroll
  for (int dt = 0; dt < 4; ++dt)
#pragma unroll
    for (int j = 0; j < 8; ++j) o[dt][j] = 0.0f;
#pragma unroll
  for (int j = 0; j < 8; ++j) { mrun[j] = -1.0e30f; lrun[j] = 0.0f; }

  const float scale = 0.125f;                        // 1/sqrt(64)
  const int krow = tid >> 2;                         // 0..31 key row
  const int kcb  = (tid & 3) * 16;                   // d col base

  for (int kc = 0; kc < sp; kc += 32) {
    // ---- cooperative K (async copy) / V-transposed loads (dilated gather) ----
    {
      size_t ntok = (size_t)b * N_ + (size_t)seg * seg_len + off +
                    (size_t)(kc + krow) * rate;
      async_copy32(&sK[krow][kcb], kb + ntok * E_ + h * D_ + kcb);
      const bf16* vp = vb + ntok * E_ + h * D_ + kcb;
#pragma unroll
      for (int e = 0; e < 16; ++e) sVT[kcb + e][krow] = vp[e];
    }
    async_drain();
    __syncthreads();

    // ---- scores S = Q K^T * scale : two 16x16 tiles over 32 keys ----
    v8f s[2];
#pragma unroll
    for (int t = 0; t < 2; ++t) {
#pragma unroll
      for (int j = 0; j < 8; ++j) s[t][j] = 0.0f;
      s[t] = wmma_bf16(qf0, load_b_frag(&sK[t * 16][0],  64 + APAD), s[t]);
      s[t] = wmma_bf16(qf1, load_b_frag(&sK[t * 16][32], 64 + APAD), s[t]);
#pragma unroll
      for (int j = 0; j < 8; ++j) s[t][j] *= scale;
    }

    // ---- online softmax: rows are striped over 16-lane halves ----
    float mnew[8], alpha[8];
#pragma unroll
    for (int j = 0; j < 8; ++j) {
      float rm = fmaxf(s[0][j], s[1][j]);
      rm = fmaxf(rm, __shfl_xor(rm, 8, 32));
      rm = fmaxf(rm, __shfl_xor(rm, 4, 32));
      rm = fmaxf(rm, __shfl_xor(rm, 2, 32));
      rm = fmaxf(rm, __shfl_xor(rm, 1, 32));
      mnew[j]  = fmaxf(mrun[j], rm);
      alpha[j] = __expf(mrun[j] - mnew[j]);
      mrun[j]  = mnew[j];
    }
#pragma unroll
    for (int dt = 0; dt < 4; ++dt)
#pragma unroll
      for (int j = 0; j < 8; ++j) o[dt][j] *= alpha[j];

#pragma unroll
    for (int j = 0; j < 8; ++j) {
      s[0][j] = __expf(s[0][j] - mnew[j]);
      s[1][j] = __expf(s[1][j] - mnew[j]);
      float rs = s[0][j] + s[1][j];
      rs += __shfl_xor(rs, 8, 32);
      rs += __shfl_xor(rs, 4, 32);
      rs += __shfl_xor(rs, 2, 32);
      rs += __shfl_xor(rs, 1, 32);
      lrun[j] = lrun[j] * alpha[j] + rs;
    }

    // ---- stash P (C layout) -> LDS, reload in A-fragment layout ----
    {
      const int half = lane >> 4;
      const int cn   = lane & 15;
#pragma unroll
      for (int t = 0; t < 2; ++t)
#pragma unroll
        for (int j = 0; j < 8; ++j)
          sP[wave][j + half * 8][t * 16 + cn] = (bf16)s[t][j];
    }
    // per-wave private stash: DS ops are in-order, just drain DScnt
    asm volatile("s_wait_dscnt 0" ::: "memory");

    v16bf pa = load_a_frag(&sP[wave][0][0], 32 + APAD);
#pragma unroll
    for (int dt = 0; dt < 4; ++dt) {
      v16bf vf = load_b_frag(&sVT[dt * 16][0], 32 + APAD);
      o[dt] = wmma_bf16(pa, vf, o[dt]);
    }
    __syncthreads();                                 // protect sK/sVT reuse
  }

  // ---- epilogue: O / l, scatter to dilated token positions ----
  const int half = lane >> 4;
  const int cn   = lane & 15;
#pragma unroll
  for (int j = 0; j < 8; ++j) {
    int q = qbase + j + half * 8;
    size_t ntok = (size_t)b * N_ + (size_t)seg * seg_len + off +
                  (size_t)q * rate;
    float inv = 1.0f / lrun[j];
#pragma unroll
    for (int dt = 0; dt < 4; ++dt)
      attn[ntok * E_ + h * D_ + dt * 16 + cn] = o[dt][j] * inv;
  }
}

// ---------------------------------------------------------------------------
// kernel_launch: weight transpose -> zero attn -> 4 projections -> 2 attention
// groups -> gated output GEMM. All on `stream`, scratch carved out of d_ws.
// ---------------------------------------------------------------------------
extern "C" void kernel_launch(void* const* d_in, const int* in_sizes, int n_in,
                              void* d_out, int out_size, void* d_ws,
                              size_t ws_size, hipStream_t stream) {
  const float* query = (const float*)d_in[0];
  const float* key_  = (const float*)d_in[1];
  const float* value = (const float*)d_in[2];
  const float* Wq = (const float*)d_in[3];  const float* bq = (const float*)d_in[4];
  const float* Wk = (const float*)d_in[5];  const float* bk = (const float*)d_in[6];
  const float* Wv = (const float*)d_in[7];  const float* bv = (const float*)d_in[8];
  const float* Wg = (const float*)d_in[9];  const float* bg = (const float*)d_in[10];
  const float* Wo = (const float*)d_in[11]; const float* bo = (const float*)d_in[12];
  float* out = (float*)d_out;

  // workspace layout
  const size_t EE2 = (size_t)E_ * E_ * sizeof(bf16);       // 2 MiB / weight
  const size_t BNE = (size_t)B_ * N_ * E_;                 // 8.39M elements
  char* ws = (char*)d_ws;
  bf16* WTq = (bf16*)(ws + 0 * EE2);
  bf16* WTk = (bf16*)(ws + 1 * EE2);
  bf16* WTv = (bf16*)(ws + 2 * EE2);
  bf16* WTg = (bf16*)(ws + 3 * EE2);
  bf16* WTo = (bf16*)(ws + 4 * EE2);
  size_t off = 5 * EE2;
  bf16*  qbuf = (bf16*)(ws + off); off += BNE * sizeof(bf16);
  bf16*  kbuf = (bf16*)(ws + off); off += BNE * sizeof(bf16);
  bf16*  vbuf = (bf16*)(ws + off); off += BNE * sizeof(bf16);
  float* gate = (float*)(ws + off); off += BNE * sizeof(float);
  float* attn = (float*)(ws + off); off += BNE * sizeof(float);
  if (ws_size < off) return;                                // ~122 MiB needed

  // 1) weights -> transposed bf16
  {
    int n = E_ * E_;
    dim3 g((n + 255) / 256), t(256);
    wconvert<<<g, t, 0, stream>>>(Wq, WTq, E_, E_);
    wconvert<<<g, t, 0, stream>>>(Wk, WTk, E_, E_);
    wconvert<<<g, t, 0, stream>>>(Wv, WTv, E_, E_);
    wconvert<<<g, t, 0, stream>>>(Wg, WTg, E_, E_);
    wconvert<<<g, t, 0, stream>>>(Wo, WTo, E_, E_);
  }
  // 2) zero attention output (dilation leaves holes that must be zero)
  {
    int n = (int)BNE;
    zerof<<<dim3((n + 255) / 256), dim3(256), 0, stream>>>(attn, n);
  }
  // 3) projections
  {
    dim3 g(M_ / BM, E_ / BN), t(256);
    gemm_bf16<<<g, t, 0, stream>>>(query, nullptr, WTq, bq, qbuf, nullptr,
                                   M_, E_, E_, 0);
    gemm_bf16<<<g, t, 0, stream>>>(key_,  nullptr, WTk, bk, kbuf, nullptr,
                                   M_, E_, E_, 0);
    gemm_bf16<<<g, t, 0, stream>>>(value, nullptr, WTv, bv, vbuf, nullptr,
                                   M_, E_, E_, 0);
    gemm_bf16<<<g, t, 0, stream>>>(query, nullptr, WTg, bg, nullptr, gate,
                                   M_, E_, E_, 1);
  }
  // 4) dilated attention, one launch per (seg_len, rate) group
  {
    // group 0: s=1024, r=1, off=0, heads 0..7, nseg=4 -> 2*4*8*16 blocks
    attn_kernel<<<dim3(B_ * 4 * 8 * 16), dim3(128), 0, stream>>>(
        qbuf, kbuf, vbuf, attn, 1024, 1, 0, 0, 4);
    // group 1: s=2048, r=2, off=1, heads 8..15, nseg=2 -> 2*2*8*16 blocks
    attn_kernel<<<dim3(B_ * 2 * 8 * 16), dim3(128), 0, stream>>>(
        qbuf, kbuf, vbuf, attn, 2048, 2, 1, 8, 2);
  }
  // 5) output projection with fused gating: out = (gate*attn) @ Wo + bo
  {
    dim3 g(M_ / BM, E_ / BN), t(256);
    gemm_bf16<<<g, t, 0, stream>>>(attn, gate, WTo, bo, nullptr, out,
                                   M_, E_, E_, 2);
  }
}
